// DistanceAwareSelfAttention_5540507812317
// MI455X (gfx1250) — matrix operand order repaired and checked by reading, not compiled, not run on hardware
//
#include <hip/hip_runtime.h>
#include <hip/hip_bf16.h>

typedef __attribute__((ext_vector_type(16))) __bf16 v16bf;
typedef __attribute__((ext_vector_type(8)))  float  v8f;
typedef __attribute__((ext_vector_type(4)))  unsigned int v4u;
typedef __attribute__((ext_vector_type(8)))  int v8i;
typedef __attribute__((ext_vector_type(4)))  int v4i;

#define SLEN 1024
#define BSZ  8
#define NH   16
#define KV   64
#define EMB  1024
#define QKV3 3072
#define NTOK (SLEN*BSZ)      // 8192
#define CLIPD 10

#if __has_builtin(__builtin_amdgcn_tensor_load_to_lds)
#define HAVE_TDM 1
#else
#define HAVE_TDM 0
#endif

__device__ __forceinline__ unsigned short f2bf(float f) {
    unsigned int u = __float_as_uint(f);
    u += 0x7FFFu + ((u >> 16) & 1u);          // round to nearest even
    return (unsigned short)(u >> 16);
}
__device__ __forceinline__ float bf2f(unsigned short h) {
    return __uint_as_float(((unsigned int)h) << 16);
}

union FragBF { uint4 q[2]; v16bf v; };

// A fragment 16x32 bf16 (M x K), row-major source with leading dim ld.
// lane L (g=L>>4): row = row0 + (L&15), K chunks [g*8, g*8+8) and [g*8+16, g*8+24)
__device__ __forceinline__ v16bf load_afrag(const unsigned short* base, int row0, int ld, int col0, int lane) {
    int g = lane >> 4;
    const unsigned short* p = base + (size_t)(row0 + (lane & 15)) * ld + col0 + g * 8;
    FragBF f;
    f.q[0] = *reinterpret_cast<const uint4*>(p);
    f.q[1] = *reinterpret_cast<const uint4*>(p + 16);
    return f.v;
}
// B fragment 32x16 bf16 (K x N). Column n of B contiguous-K in memory:
// memory row (n0 + lane&15), K chunks [g*8, g*8+8) and [g*8+16, g*8+24) at col k0, g=lane>>4.
__device__ __forceinline__ v16bf load_bfrag(const unsigned short* base, int n0, int ld, int k0, int lane) {
    const unsigned short* p = base + (size_t)(n0 + (lane & 15)) * ld + k0 + (lane >> 4) * 8;
    FragBF f;
    f.q[0] = *reinterpret_cast<const uint4*>(p);
    f.q[1] = *reinterpret_cast<const uint4*>(p + 16);
    return f.v;
}
// A fragment from a 16x32 row-major LDS tile (ld = 32)
__device__ __forceinline__ v16bf load_afrag_lds(const unsigned short* base, int lane) {
    int g = lane >> 4;
    const unsigned short* p = base + (lane & 15) * 32 + g * 8;
    FragBF f;
    f.q[0] = *reinterpret_cast<const uint4*>(p);
    f.q[1] = *reinterpret_cast<const uint4*>(p + 16);
    return f.v;
}

// ---- Tensor Data Mover: DMA a 32x64 bf16 tile (rows of a (SLEN x 64) row-major tensor) to LDS ----
__device__ __forceinline__ void tdm_load_k_tile(unsigned int lds_byte_addr, const unsigned short* gtile) {
#if HAVE_TDM
    unsigned long long ga = (unsigned long long)(size_t)gtile;
    v4u g0;
    g0.x = 1u;                                   // count=1, is_restore=0, gather off
    g0.y = lds_byte_addr;                        // D#.lds_addr
    g0.z = (unsigned int)ga;                     // global_addr[31:0]
    g0.w = (unsigned int)((ga >> 32) & 0x01FFFFFFull) | (2u << 30);   // addr[56:32] | type=2
    v8i g1 = {};
    g1[0] = (int)(1u << 16);                     // workgroup_mask=0, data_size=1 (2 bytes)
    g1[1] = (int)(KV << 16);                     // tensor_dim0[15:0]=64  (bits 63:48)
    g1[2] = (int)(SLEN << 16);                   // tensor_dim1[15:0]=1024 (bits 95:80)
    g1[3] = (int)(KV << 16);                     // tile_dim0=64 (bits 127:112)
    g1[4] = 32;                                  // tile_dim1=32 (bits 143:128)
    g1[5] = KV;                                  // tensor_dim0_stride=64 (bits 207:160)
    v4i z4 = {};
#if defined(__clang_major__) && (__clang_major__ >= 23)
    v8i z8 = {};
    __builtin_amdgcn_tensor_load_to_lds(g0, g1, z4, z4, z8, 0);
#else
    __builtin_amdgcn_tensor_load_to_lds(g0, g1, z4, z4, 0);
#endif
#endif
}

__global__ void __launch_bounds__(256) cvt_bf16_kernel(const float* __restrict__ src,
                                                       unsigned short* __restrict__ dst, int n) {
    int i = blockIdx.x * blockDim.x + threadIdx.x;
    int stride = gridDim.x * blockDim.x;
    for (; i < n; i += stride) dst[i] = f2bf(src[i]);
}

// ---------------- QKV GEMM: (8192 x 1024) @ (1024 x 3072)^T, route to Q/K/Vt ----------------
// Each wave: 32(M) x 64(N) tile -> 8 WMMA per 32-wide K step, A/B fragment reuse.
__global__ void __launch_bounds__(256) qkv_gemm_kernel(const unsigned short* __restrict__ Xbf,
                                                       const unsigned short* __restrict__ Wbf,
                                                       const float* __restrict__ b_in,
                                                       unsigned short* __restrict__ Q,
                                                       unsigned short* __restrict__ Kb,
                                                       unsigned short* __restrict__ Vt) {
    int wave = threadIdx.x >> 5, lane = threadIdx.x & 31;
    int n0 = blockIdx.x * 256 + wave * 32;    // token rows (32 per wave)
    int m0 = blockIdx.y * 64;                 // output feature cols (64 per wave)
    v8f c[2][4] = {};
    #pragma unroll 2
    for (int k = 0; k < EMB; k += 32) {
        v16bf a0 = load_afrag(Xbf, n0,      EMB, k, lane);
        v16bf a1 = load_afrag(Xbf, n0 + 16, EMB, k, lane);
        #pragma unroll
        for (int jc = 0; jc < 4; jc++) {
            v16bf b = load_bfrag(Wbf, m0 + jc * 16, EMB, k, lane);
            c[0][jc] = __builtin_amdgcn_wmma_f32_16x16x32_bf16(false, a0, false, b, (short)0, c[0][jc], false, false);
            c[1][jc] = __builtin_amdgcn_wmma_f32_16x16x32_bf16(false, a1, false, b, (short)0, c[1][jc], false, false);
        }
    }
    int N = lane & 15, Mo = (lane >> 4) * 8;
    int sel = m0 >> 10;                 // uniform: 0=q 1=k 2=v
    int hh  = (m0 >> 6) & (NH - 1);     // uniform per wave tile
    #pragma unroll
    for (int jr = 0; jr < 2; jr++) {
        #pragma unroll
        for (int jc = 0; jc < 4; jc++) {
            #pragma unroll
            for (int r = 0; r < 8; r++) {
                int n = n0 + jr * 16 + Mo + r;
                int m = m0 + jc * 16 + N;
                float val = c[jr][jc][r] + b_in[m];
                int s = n >> 3, bb = n & 7;
                int dd = jc * 16 + N;           // m0 is 64-aligned
                size_t bh = (size_t)bb * NH + hh;
                if (sel == 0) {
                    Q[(bh * SLEN + s) * KV + dd] = f2bf(val * 0.125f);   // d^-0.5 = 1/8
                } else if (sel == 1) {
                    Kb[(bh * SLEN + s) * KV + dd] = f2bf(val);
                } else {
                    Vt[(bh * KV + dd) * SLEN + s] = f2bf(val);           // V transposed (d, s)
                }
            }
        }
    }
}

// ---------------- Fused flash attention (one wave = 16 query rows of one (b,h)) ----------------
// K tiles (32x64 bf16) are staged into LDS by the Tensor Data Mover, double buffered.
__global__ void __launch_bounds__(256) attn_kernel(const unsigned short* __restrict__ Q,
                                                   const unsigned short* __restrict__ Kb,
                                                   const unsigned short* __restrict__ Vt,
                                                   const float* __restrict__ kemb,
                                                   const float* __restrict__ vemb,
                                                   unsigned short* __restrict__ Abuf) {
    __shared__ float s_kemb[(CLIPD + 1) * KV];
    __shared__ float s_vemb[(CLIPD + 1) * KV];
    __shared__ float s_coef[8][16][CLIPD + 1];
    __shared__ unsigned short s_p[8][16][32];
    __shared__ float s_cap[8][2][16][32];
    extern __shared__ unsigned short kdyn[];      // 8 waves * 2 bufs * (32*64) bf16 = 64 KB

    const int KTILE = 32 * KV;                    // elements per K tile

    int tid = threadIdx.x, wave = tid >> 5, lane = tid & 31;
    for (int i = tid; i < (CLIPD + 1) * KV; i += 256) { s_kemb[i] = kemb[i]; s_vemb[i] = vemb[i]; }
    __syncthreads();

    int w = blockIdx.x * 8 + wave;                // 8192 wave tasks
    int sblk = w & 63;
    int hh = (w >> 6) & (NH - 1);
    int bb = w >> 10;
    int s0 = sblk * 16;
    size_t bh = (size_t)bb * NH + hh;
    const unsigned short* Qp = Q  + bh * SLEN * KV;
    const unsigned short* Kp = Kb + bh * SLEN * KV;
    const unsigned short* Vp = Vt + bh * KV * SLEN;

    unsigned short* kw0 = kdyn + wave * 2 * KTILE;
    unsigned int kw0_lds = (unsigned int)(size_t)kw0;

    // prologue: start DMA of K tile 0
#if HAVE_TDM
    tdm_load_k_tile(kw0_lds, Kp);
#else
    for (int i = lane; i < KTILE / 8; i += 32)
        *reinterpret_cast<uint4*>(kw0 + i * 8) = *reinterpret_cast<const uint4*>(Kp + i * 8);
#endif

    // bias coefficients: coef[row][c] = dot(q_scaled[row], k_emb[c])
    for (int i = lane; i < 16 * (CLIPD + 1); i += 32) {
        int r = i / (CLIPD + 1), cc = i % (CLIPD + 1);
        const unsigned short* qr = Qp + (size_t)(s0 + r) * KV;
        float acc = 0.f;
        for (int d = 0; d < KV; d++) acc += bf2f(qr[d]) * s_kemb[cc * KV + d];
        s_coef[wave][r][cc] = acc;
    }

    v16bf aq0 = load_afrag(Qp, s0, KV, 0, lane);
    v16bf aq1 = load_afrag(Qp, s0, KV, 32, lane);

    float mrow[8], lrow[8];
    #pragma unroll
    for (int r = 0; r < 8; r++) { mrow[r] = -1e30f; lrow[r] = 0.f; }
    v8f o0 = {}, o1 = {}, o2 = {}, o3 = {};

    int lo = s0 - 9; if (lo < 0) lo = 0;
    int hi = s0 + 24; if (hi > SLEN - 1) hi = SLEN - 1;
    int cb0 = lo >> 5, cb1 = hi >> 5;

    int N = lane & 15, Mo = (lane >> 4) * 8;

    asm volatile("s_wait_dscnt 0" ::: "memory");   // coef + prologue copy visible wave-wide

    for (int tb = 0; tb < SLEN / 32; ++tb) {
        int t0 = tb * 32;
        unsigned short* kw = kw0 + (tb & 1) * KTILE;
        // kick DMA for next tile into the other buffer (its last readers finished
        // before the s_wait_dscnt of the previous iteration)
        if (tb + 1 < SLEN / 32) {
#if HAVE_TDM
            tdm_load_k_tile(kw0_lds + (unsigned int)(((tb + 1) & 1) * KTILE * 2),
                            Kp + (size_t)(tb + 1) * KTILE);
            __builtin_amdgcn_s_wait_tensorcnt((short)1);   // current tile done, next in flight
#else
            unsigned short* kn = kw0 + ((tb + 1) & 1) * KTILE;
            const unsigned short* gp = Kp + (size_t)(tb + 1) * KTILE;
            for (int i = lane; i < KTILE / 8; i += 32)
                *reinterpret_cast<uint4*>(kn + i * 8) = *reinterpret_cast<const uint4*>(gp + i * 8);
            asm volatile("s_wait_dscnt 0" ::: "memory");
#endif
        } else {
#if HAVE_TDM
            __builtin_amdgcn_s_wait_tensorcnt((short)0);
#endif
        }
        asm volatile("" ::: "memory");

        // ---- S = Qs @ K^T  (16 x 32), f32 acc; K from LDS ----
        v16bf bk;
        v8f sA = {}, sB = {};
        bk = load_bfrag(kw, 0,  KV, 0,  lane);
        sA = __builtin_amdgcn_wmma_f32_16x16x32_bf16(false, aq0, false, bk, (short)0, sA, false, false);
        bk = load_bfrag(kw, 0,  KV, 32, lane);
        sA = __builtin_amdgcn_wmma_f32_16x16x32_bf16(false, aq1, false, bk, (short)0, sA, false, false);
        bk = load_bfrag(kw, 16, KV, 0,  lane);
        sB = __builtin_amdgcn_wmma_f32_16x16x32_bf16(false, aq0, false, bk, (short)0, sB, false, false);
        bk = load_bfrag(kw, 16, KV, 32, lane);
        sB = __builtin_amdgcn_wmma_f32_16x16x32_bf16(false, aq1, false, bk, (short)0, sB, false, false);

        // ---- add distance bias ----
        #pragma unroll
        for (int r = 0; r < 8; r++) {
            int M = Mo + r, s = s0 + M;
            int ta = t0 + N, tbc = t0 + 16 + N;
            int da = s - ta; if (da < 0) da = -da; if (da > CLIPD) da = CLIPD;
            int db = s - tbc; if (db < 0) db = -db; if (db > CLIPD) db = CLIPD;
            sA[r] += s_coef[wave][M][da];
            sB[r] += s_coef[wave][M][db];
        }
        // ---- capture raw scores of diagonal-band blocks ----
        if (tb == cb0 || tb == cb1) {
            int slot = (tb == cb0) ? 0 : 1;
            #pragma unroll
            for (int r = 0; r < 8; r++) {
                s_cap[wave][slot][Mo + r][N]      = sA[r];
                s_cap[wave][slot][Mo + r][16 + N] = sB[r];
            }
        }
        // ---- online softmax + P to LDS (bf16) ----
        #pragma unroll
        for (int r = 0; r < 8; r++) {
            float v = fmaxf(sA[r], sB[r]);
            v = fmaxf(v, __shfl_xor(v, 1));
            v = fmaxf(v, __shfl_xor(v, 2));
            v = fmaxf(v, __shfl_xor(v, 4));
            v = fmaxf(v, __shfl_xor(v, 8));
            float mn = fmaxf(mrow[r], v);
            float alpha = __expf(mrow[r] - mn);
            float p0 = __expf(sA[r] - mn);
            float p1 = __expf(sB[r] - mn);
            float rs = p0 + p1;
            rs += __shfl_xor(rs, 1);
            rs += __shfl_xor(rs, 2);
            rs += __shfl_xor(rs, 4);
            rs += __shfl_xor(rs, 8);
            lrow[r] = lrow[r] * alpha + rs;
            mrow[r] = mn;
            o0[r] *= alpha; o1[r] *= alpha; o2[r] *= alpha; o3[r] *= alpha;
            s_p[wave][Mo + r][N]      = f2bf(p0);
            s_p[wave][Mo + r][16 + N] = f2bf(p1);
        }
        asm volatile("s_wait_dscnt 0" ::: "memory");
        v16bf pa = load_afrag_lds(&s_p[wave][0][0], lane);
        // ---- O += P @ V  (V stored transposed: rows=d, cols=t) ----
        v16bf bv;
        bv = load_bfrag(Vp, 0,  SLEN, t0, lane);
        o0 = __builtin_amdgcn_wmma_f32_16x16x32_bf16(false, pa, false, bv, (short)0, o0, false, false);
        bv = load_bfrag(Vp, 16, SLEN, t0, lane);
        o1 = __builtin_amdgcn_wmma_f32_16x16x32_bf16(false, pa, false, bv, (short)0, o1, false, false);
        bv = load_bfrag(Vp, 32, SLEN, t0, lane);
        o2 = __builtin_amdgcn_wmma_f32_16x16x32_bf16(false, pa, false, bv, (short)0, o2, false, false);
        bv = load_bfrag(Vp, 48, SLEN, t0, lane);
        o3 = __builtin_amdgcn_wmma_f32_16x16x32_bf16(false, pa, false, bv, (short)0, o3, false, false);
    }

    asm volatile("s_wait_dscnt 0" ::: "memory");
    // ---- epilogue: normalize + v_dist correction from diagonal band ----
    #pragma unroll
    for (int r = 0; r < 8; r++) {
        int M = Mo + r, s = s0 + M;
        float invl = 1.0f / lrow[r];
        float pj[19];
        #pragma unroll
        for (int j = -9; j <= 9; j++) {
            int t = s + j;
            float p = 0.f;
            if (t >= 0 && t < SLEN) {
                int blk = t >> 5;
                int slot = (blk == cb0) ? 0 : ((blk == cb1) ? 1 : -1);
                if (slot >= 0) p = __expf(s_cap[wave][slot][M][t & 31] - mrow[r]) * invl;
            }
            pj[j + 9] = p;
        }
        #pragma unroll
        for (int db = 0; db < 4; db++) {
            int dcol = db * 16 + N;
            float v10 = s_vemb[CLIPD * KV + dcol];
            float corr = v10;
            #pragma unroll
            for (int jj = 0; jj < 19; jj++) {
                int cc = (jj < 9) ? (9 - jj) : (jj - 9);
                corr += pj[jj] * (s_vemb[cc * KV + dcol] - v10);
            }
            float oacc = (db == 0) ? o0[r] : (db == 1) ? o1[r] : (db == 2) ? o2[r] : o3[r];
            float val = oacc * invl + corr;
            Abuf[((size_t)s * BSZ + bb) * (NH * KV) + hh * KV + dcol] = f2bf(val);
        }
    }
}

// ---------------- Output projection: (8192 x 1024) @ (1024 x 1024)^T + b_out -> fp32 ----------------
__global__ void __launch_bounds__(256) out_gemm_kernel(const unsigned short* __restrict__ A,
                                                       const unsigned short* __restrict__ W,
                                                       const float* __restrict__ bias,
                                                       float* __restrict__ out) {
    int wave = threadIdx.x >> 5, lane = threadIdx.x & 31;
    int n0 = blockIdx.x * 256 + wave * 32;
    int e0 = blockIdx.y * 64;
    v8f c[2][4] = {};
    #pragma unroll 2
    for (int k = 0; k < EMB; k += 32) {
        v16bf a0 = load_afrag(A, n0,      EMB, k, lane);
        v16bf a1 = load_afrag(A, n0 + 16, EMB, k, lane);
        #pragma unroll
        for (int jc = 0; jc < 4; jc++) {
            v16bf b = load_bfrag(W, e0 + jc * 16, EMB, k, lane);
            c[0][jc] = __builtin_amdgcn_wmma_f32_16x16x32_bf16(false, a0, false, b, (short)0, c[0][jc], false, false);
            c[1][jc] = __builtin_amdgcn_wmma_f32_16x16x32_bf16(false, a1, false, b, (short)0, c[1][jc], false, false);
        }
    }
    int N = lane & 15, Mo = (lane >> 4) * 8;
    #pragma unroll
    for (int jr = 0; jr < 2; jr++) {
        #pragma unroll
        for (int jc = 0; jc < 4; jc++) {
            #pragma unroll
            for (int r = 0; r < 8; r++) {
                int n = n0 + jr * 16 + Mo + r, e = e0 + jc * 16 + N;
                out[(size_t)n * EMB + e] = c[jr][jc][r] + bias[e];
            }
        }
    }
}

extern "C" void kernel_launch(void* const* d_in, const int* in_sizes, int n_in,
                              void* d_out, int out_size, void* d_ws, size_t ws_size,
                              hipStream_t stream) {
    const float* X     = (const float*)d_in[0];   // (1024, 8, 1024)
    const float* W_in  = (const float*)d_in[1];   // (3072, 1024)
    const float* b_in  = (const float*)d_in[2];   // (3072,)
    const float* kemb  = (const float*)d_in[3];   // (11, 64)
    const float* vemb  = (const float*)d_in[4];   // (11, 64)
    const float* W_out = (const float*)d_in[5];   // (1024, 1024)
    const float* b_out = (const float*)d_in[6];   // (1024,)

    char* ws = (char*)d_ws;
    size_t off = 0;
    unsigned short* Xbf    = (unsigned short*)(ws + off); off += (size_t)NTOK * EMB * 2;
    unsigned short* Winbf  = (unsigned short*)(ws + off); off += (size_t)QKV3 * EMB * 2;
    unsigned short* Woutbf = (unsigned short*)(ws + off); off += (size_t)EMB * EMB * 2;
    unsigned short* Qb     = (unsigned short*)(ws + off); off += (size_t)BSZ * NH * SLEN * KV * 2;
    unsigned short* Kb     = (unsigned short*)(ws + off); off += (size_t)BSZ * NH * SLEN * KV * 2;
    unsigned short* Vt     = (unsigned short*)(ws + off); off += (size_t)BSZ * NH * KV * SLEN * 2;
    unsigned short* Abuf   = (unsigned short*)(ws + off); off += (size_t)NTOK * (NH * KV) * 2;

    cvt_bf16_kernel<<<1024, 256, 0, stream>>>(X, Xbf, NTOK * EMB);
    cvt_bf16_kernel<<<512,  256, 0, stream>>>(W_in, Winbf, QKV3 * EMB);
    cvt_bf16_kernel<<<256,  256, 0, stream>>>(W_out, Woutbf, EMB * EMB);

    qkv_gemm_kernel<<<dim3(NTOK / 256, QKV3 / 64), 256, 0, stream>>>(Xbf, Winbf, b_in, Qb, Kb, Vt);

    // 64 KB dynamic LDS: 8 waves * double-buffered 32x64 bf16 K tiles
    attn_kernel<<<(BSZ * NH * (SLEN / 16)) / 8, 256, 8 * 2 * 32 * KV * 2, stream>>>(
        Qb, Kb, Vt, kemb, vemb, Abuf);

    out_gemm_kernel<<<dim3(NTOK / 256, EMB / 64), 256, 0, stream>>>(Abuf, Woutbf, b_out, (float*)d_out);
}
